// TransformerDecoder_3539053052152
// MI455X (gfx1250) — matrix operand order repaired and checked by reading, not compiled, not run on hardware
//
#include <hip/hip_runtime.h>
#include <cstdint>
#include <cstddef>

// ---------------- problem constants ----------------
static constexpr int BB    = 128;
static constexpr int HH    = 1024;
static constexpr int STYLE = 320;
static constexpr int NMEL  = 80;
static constexpr int TT    = 2000;
static constexpr int SEQ   = 400;

// GEMM1: X1 row = [txt 0..1023 | style 1024..1343 | prev 1344..1423 | pad 1424..1439 | h1 1440..2463]
static constexpr int ROW1  = 2464;          // padded K1 (77 * 32)
static constexpr int KT1   = 77;
static constexpr int ROW1B = ROW1 * 2;      // bytes per X1 row (bf16)
// GEMM2: X2 row = [h1 0..1023 | h2 1024..2047]
static constexpr int ROW2  = 2048;          // K2 (64 * 32)
static constexpr int KT2   = 64;
static constexpr int ROW2B = ROW2 * 2;

static constexpr int NWG = 128;             // persistent workgroups (assume >=128 WGPs)
static constexpr int NG1 = 64;              // group-1 (GEMM1) WGs; group-2 = 64..127

static constexpr int X1BUF = BB * ROW1;     // elements per X1 buffer (x2 buffers)
static constexpr int X2BUF = BB * ROW2;

static constexpr int OFF_STYLE = 1024;
static constexpr int OFF_PREV  = 1344;
static constexpr int OFF_H1    = 1440;

// swizzled weight sizes (dwords; each dword = 2 bf16)
static constexpr int W1DW = NG1 * 4 * KT1 * 256;   // 5,046,272 dwords (~20.2 MB)
static constexpr int W2DW = NG1 * 4 * KT2 * 256;   // 4,194,304 dwords (~16.8 MB)
static constexpr int FCDW = 5 * 32 * 256;          // 40,960 dwords
static constexpr int LDS_BYTES = 4 * KT1 * 1024;   // 315,392 B (group-1 worst case)

typedef __bf16       v16bf __attribute__((ext_vector_type(16)));
typedef float        v8f   __attribute__((ext_vector_type(8)));
typedef unsigned int v4u   __attribute__((ext_vector_type(4)));

union FragAB { v4u u[2]; v16bf bf; };
union BFBits { __bf16 h; unsigned short s; };

__device__ __forceinline__ v8f splat8(float x) {
  v8f v;
#pragma unroll
  for (int i = 0; i < 8; ++i) v[i] = x;
  return v;
}

// ---- branch-free transcendentals on the recurrence critical path ----
__device__ __forceinline__ float fast_exp2(float x) {
#if __has_builtin(__builtin_amdgcn_exp2f)
  return __builtin_amdgcn_exp2f(x);        // raw v_exp_f32
#else
  return exp2f(x);
#endif
}
__device__ __forceinline__ float fast_rcp(float x) {
#if __has_builtin(__builtin_amdgcn_rcpf)
  return __builtin_amdgcn_rcpf(x);         // raw v_rcp_f32
#else
  return 1.0f / x;
#endif
}
__device__ __forceinline__ float sigm(float x) {
  // 1 / (1 + 2^(-x*log2(e))) : saturates via exp2->inf->rcp->0, no branches
  return fast_rcp(1.0f + fast_exp2(-1.44269504f * x));
}
__device__ __forceinline__ float fast_tanh(float x) {
#if __has_builtin(__builtin_amdgcn_tanhf)
  return __builtin_amdgcn_tanhf(x);        // gfx1250 hardware v_tanh_f32
#else
  // tanh(x) = 1 - 2/(e^(2x)+1); exp2 overflow -> inf -> rcp -> 0 -> +1, underflow -> -1
  float e = fast_exp2(2.88539008f * x);
  return 1.0f - 2.0f * fast_rcp(e + 1.0f);
#endif
}

// ---------------- grid-wide sense barrier (persistent kernel) ----------------
__device__ __forceinline__ void grid_barrier(int* bar) {
  __threadfence();
  __syncthreads();
  if (threadIdx.x == 0) {
    int g = __hip_atomic_load(bar + 1, __ATOMIC_RELAXED, __HIP_MEMORY_SCOPE_AGENT);
    int t = __hip_atomic_fetch_add(bar, 1, __ATOMIC_ACQ_REL, __HIP_MEMORY_SCOPE_AGENT);
    if (t == (int)gridDim.x - 1) {
      __hip_atomic_store(bar, 0, __ATOMIC_RELAXED, __HIP_MEMORY_SCOPE_AGENT);
      __hip_atomic_store(bar + 1, g + 1, __ATOMIC_RELEASE, __HIP_MEMORY_SCOPE_AGENT);
    } else {
      while (__hip_atomic_load(bar + 1, __ATOMIC_ACQUIRE, __HIP_MEMORY_SCOPE_AGENT) == g)
        __builtin_amdgcn_s_sleep(2);
    }
  }
  __syncthreads();
  __threadfence();
}

// ---------------- weight swizzle prep kernels ----------------
// Layout per 32x16 bf16 B-tile: 32 lanes x 8 dwords (lane L: col n = tile_n*16 + L%16,
// dword v holds K = kt*32 + (L/16)*16 + 2v .. +1).  Flat: ((((q*4+g)*KT)+kt)*32 + lane)*8 + v.
__global__ __launch_bounds__(256) void prep_w1(const float* __restrict__ W_ih1,
                                               const float* __restrict__ W_hh1,
                                               unsigned int* __restrict__ W1sw) {
  long id = (long)blockIdx.x * 256 + threadIdx.x;
  if (id >= (long)W1DW) return;
  int v = id & 7, l = (id >> 3) & 31;
  long rest = id >> 8;
  int kt = (int)(rest % KT1);
  long rest2 = rest / KT1;
  int g = (int)(rest2 & 3), q = (int)(rest2 >> 2);
  int n  = g * 1024 + q * 16 + (l & 15);
  int k0 = kt * 32 + (l >> 4) * 16 + v * 2;
  unsigned int pack = 0;
#pragma unroll
  for (int p = 0; p < 2; ++p) {
    int kk = k0 + p;
    float val;
    if (kk < 1424)       val = W_ih1[(size_t)n * 1424 + kk];       // [txt|style|prev]
    else if (kk < 1440)  val = 0.0f;                               // pad
    else                 val = W_hh1[(size_t)n * 1024 + (kk - 1440)];
    BFBits cv; cv.h = (__bf16)val;
    pack |= ((unsigned int)cv.s) << (16 * p);
  }
  W1sw[id] = pack;
}

__global__ __launch_bounds__(256) void prep_w2(const float* __restrict__ W_ih2,
                                               const float* __restrict__ W_hh2,
                                               unsigned int* __restrict__ W2sw) {
  long id = (long)blockIdx.x * 256 + threadIdx.x;
  if (id >= (long)W2DW) return;
  int v = id & 7, l = (id >> 3) & 31;
  long rest = id >> 8;
  int kt = (int)(rest & 63);
  long rest2 = rest >> 6;
  int g = (int)(rest2 & 3), q = (int)(rest2 >> 2);
  int n  = g * 1024 + q * 16 + (l & 15);
  int k0 = kt * 32 + (l >> 4) * 16 + v * 2;
  unsigned int pack = 0;
#pragma unroll
  for (int p = 0; p < 2; ++p) {
    int kk = k0 + p;
    float val = (kk < 1024) ? W_ih2[(size_t)n * 1024 + kk]
                            : W_hh2[(size_t)n * 1024 + (kk - 1024)];
    BFBits cv; cv.h = (__bf16)val;
    pack |= ((unsigned int)cv.s) << (16 * p);
  }
  W2sw[id] = pack;
}

__global__ __launch_bounds__(256) void prep_fc(const float* __restrict__ fc_w,
                                               unsigned int* __restrict__ FCsw) {
  int id = blockIdx.x * 256 + threadIdx.x;
  if (id >= FCDW) return;
  int v = id & 7, l = (id >> 3) & 31;
  int rest = id >> 8;
  int kt = rest & 31, mt = rest >> 5;
  int m  = mt * 16 + (l & 15);
  int k0 = kt * 32 + (l >> 4) * 16 + v * 2;
  unsigned int pack = 0;
#pragma unroll
  for (int p = 0; p < 2; ++p) {
    BFBits cv; cv.h = (__bf16)fc_w[(size_t)m * 1024 + (k0 + p)];
    pack |= ((unsigned int)cv.s) << (16 * p);
  }
  FCsw[id] = pack;
}

// ---------------- FC tile: frame[16b x 16m] = h2 . fc_w^T + fc_b ----------------
__device__ __forceinline__ void fc_tile(int job, int tframe, const __bf16* X2buf,
                                        const unsigned int* FCsw, const float* fc_b,
                                        float* out, int lane) {
  const int half = lane >> 4, lm = lane & 15;
  const int btile = job % 8, mtile = job / 8;       // 8 B-tiles x 5 mel-tiles
  const int mcol = mtile * 16 + lm;
  v8f acc = splat8(fc_b[mcol]);
  const char* arow = (const char*)X2buf + (size_t)(btile * 16 + lm) * ROW2B + 2048; // h2 half
  const char* wbase = (const char*)FCsw;
  for (int kt = 0; kt < 32; ++kt) {
    FragAB a, b;
    a.u[0] = *(const v4u*)(arow + kt * 64 + half * 32);
    a.u[1] = *(const v4u*)(arow + kt * 64 + half * 32 + 16);
    const char* bp = wbase + ((size_t)(mtile * 32 + kt) * 32 + lane) * 32;
    b.u[0] = *(const v4u*)bp;
    b.u[1] = *(const v4u*)(bp + 16);
    acc = __builtin_amdgcn_wmma_f32_16x16x32_bf16(false, a.bf, false, b.bf,
                                                  (short)0, acc, false, false);
  }
#pragma unroll
  for (int r = 0; r < 8; ++r) {
    int b = btile * 16 + r + half * 8;
    out[((size_t)b * NMEL + mcol) * TT + tframe] = acc[r];
  }
}

// ---------------- persistent decoder kernel ----------------
// WGs 0..63  : GEMM1 quad-tiles (gate cols q*16..q*16+15 for i,f,g,o), weights in LDS (308KB)
// WGs 64..127: GEMM2 quad-tiles, weights in LDS (256KB); also FC(t-1) + input assembly in P1
__global__ __launch_bounds__(256)
void decoder_persistent(const float* __restrict__ enc_out, const float* __restrict__ style,
                        const float* __restrict__ mel,
                        const float* __restrict__ b_ih1, const float* __restrict__ b_hh1,
                        const float* __restrict__ b_ih2, const float* __restrict__ b_hh2,
                        const float* __restrict__ fc_b,
                        const unsigned int* __restrict__ W1sw,
                        const unsigned int* __restrict__ W2sw,
                        const unsigned int* __restrict__ FCsw,
                        __bf16* __restrict__ X1, __bf16* __restrict__ X2,
                        int* __restrict__ bar, float* __restrict__ out) {
  extern __shared__ char smem[];
  const int w    = blockIdx.x;
  const int tid  = threadIdx.x;
  const int lane = tid & 31;
  const int wave = tid >> 5;        // 8 waves: wave = B-tile (rows wave*16..wave*16+15)
  const int half = lane >> 4, lm = lane & 15;
  const bool isG1 = (w < NG1);
  const int q = isG1 ? w : (w - NG1);          // gate column quad 0..63
  const int hcol = q * 16 + lm;                // column within H

  // ---- load this WG's swizzled weights into LDS (once) ----
  {
    v4u* dst = (v4u*)smem;
    if (isG1) {
      const v4u* src = (const v4u*)(W1sw + (size_t)w * (4 * KT1 * 256));
      for (int i = tid; i < 4 * KT1 * 64; i += 256) dst[i] = src[i];
    } else {
      const v4u* src = (const v4u*)(W2sw + (size_t)(w - NG1) * (4 * KT2 * 256));
      for (int i = tid; i < 4 * KT2 * 64; i += 256) dst[i] = src[i];
    }
  }

  // ---- cooperative init of activation buffers ----
  const int gtid = w * 256 + tid, gstride = NWG * 256;
  {
    unsigned int* X1u = (unsigned int*)X1;     // 2 buffers = X1BUF dwords total
    for (int i = gtid; i < X1BUF; i += gstride) X1u[i] = 0u;
    unsigned int* X2u = (unsigned int*)X2;     // 2 buffers = X2BUF dwords total
    for (int i = gtid; i < X2BUF; i += gstride) X2u[i] = 0u;
  }
  grid_barrier(bar);
  {
    for (int i = gtid; i < BB * STYLE; i += gstride) {   // style (constant, both buffers)
      int b = i / STYLE, s = i - b * STYLE;
      __bf16 v = (__bf16)style[(size_t)b * STYLE + s];
      X1[(size_t)b * ROW1 + OFF_STYLE + s] = v;
      X1[(size_t)X1BUF + (size_t)b * ROW1 + OFF_STYLE + s] = v;
    }
    for (int i = gtid; i < BB * HH; i += gstride) {      // txt(0) = enc_out[:,0,:]
      int b = i >> 10, k = i & 1023;
      X1[(size_t)b * ROW1 + k] = (__bf16)enc_out[(size_t)b * SEQ * HH + k];
    }
    // prev(0)=0, h1(0)=0, h2(-1)=0 already from zeroing
  }
  grid_barrier(bar);

  // per-wave constant biases for this quad-tile
  float bias[4];
#pragma unroll
  for (int g = 0; g < 4; ++g)
    bias[g] = isG1 ? (b_ih1[g * 1024 + hcol] + b_hh1[g * 1024 + hcol])
                   : (b_ih2[g * 1024 + hcol] + b_hh2[g * 1024 + hcol]);

  v8f creg = splat8(0.0f);   // c1 (G1 waves) / c2 (G2 waves) live in VGPRs for all 2000 steps

#pragma unroll 1
  for (int t = 0; t < TT; ++t) {
    __bf16* X1cur = X1 + (size_t)(t & 1) * X1BUF;
    __bf16* X1nxt = X1 + (size_t)((t + 1) & 1) * X1BUF;
    __bf16* X2cur = X2 + (size_t)(t & 1) * X2BUF;
    __bf16* X2nxt = X2 + (size_t)((t + 1) & 1) * X2BUF;

    // ================= P1 =================
    if (isG1) {
      // gates1 tile: rows wave*16..+15, cols {hcol, +1024, +2048, +3072}; K = 2464
      v8f acc[4];
#pragma unroll
      for (int g = 0; g < 4; ++g) acc[g] = splat8(bias[g]);
      const char* arow = (const char*)X1cur + (size_t)(wave * 16 + lm) * ROW1B;
#pragma unroll 1
      for (int kt = 0; kt < KT1; ++kt) {
        FragAB a;
        a.u[0] = *(const v4u*)(arow + kt * 64 + half * 32);
        a.u[1] = *(const v4u*)(arow + kt * 64 + half * 32 + 16);
#pragma unroll
        for (int g = 0; g < 4; ++g) {
          FragAB b;
          const char* bp = smem + (((g * KT1 + kt) << 10) + lane * 32);
          b.u[0] = *(const v4u*)bp;
          b.u[1] = *(const v4u*)(bp + 16);
          acc[g] = __builtin_amdgcn_wmma_f32_16x16x32_bf16(false, a.bf, false, b.bf,
                                                           (short)0, acc[g], false, false);
        }
      }
      // LSTM1 pointwise (branch-free); h1 -> X2cur[:,0:1024] and X1nxt h1-region
#pragma unroll
      for (int r = 0; r < 8; ++r) {
        float i_ = sigm(acc[0][r]), f_ = sigm(acc[1][r]);
        float g_ = fast_tanh(acc[2][r]), o_ = sigm(acc[3][r]);
        float cn = f_ * creg[r] + i_ * g_;
        creg[r] = cn;
        __bf16 hb = (__bf16)(o_ * fast_tanh(cn));
        int b = wave * 16 + r + half * 8;
        X2cur[(size_t)b * ROW2 + hcol] = hb;
        X1nxt[(size_t)b * ROW1 + OFF_H1 + hcol] = hb;
      }
    } else {
      // FC(t-1): 40 tile jobs on wave 0 of WGs 64..103 (reads h2(t-1) in X2cur[:,1024:])
      const int w2 = w - NG1;
      if (t > 0 && w2 < 40 && wave == 0)
        fc_tile(w2, t - 1, X2cur, FCsw, fc_b, out, lane);
      // assemble txt/prev for step t+1 into X1nxt (disjoint from the h1 region G1 writes)
      if (t + 1 < TT) {
        const int g2 = w2 * 256 + tid, g2s = NG1 * 256;
        const int it = (t + 1 < SEQ) ? (t + 1) : (SEQ - 1);
        for (int i = g2; i < BB * HH; i += g2s) {
          int b = i >> 10, k = i & 1023;
          X1nxt[(size_t)b * ROW1 + k] = (__bf16)enc_out[(size_t)b * SEQ * HH + (size_t)it * HH + k];
        }
        for (int i = g2; i < BB * NMEL; i += g2s) {      // prev(t+1) = mel[:, :, t]
          int b = i / NMEL, m = i - b * NMEL;
          X1nxt[(size_t)b * ROW1 + OFF_PREV + m] = (__bf16)mel[((size_t)b * NMEL + m) * TT + t];
        }
      }
    }
    grid_barrier(bar);

    // ================= P2 =================
    if (!isG1) {
      // gates2 tile; K = 2048 over X2cur rows [h1 | h2(t-1)]
      v8f acc[4];
#pragma unroll
      for (int g = 0; g < 4; ++g) acc[g] = splat8(bias[g]);
      const char* arow = (const char*)X2cur + (size_t)(wave * 16 + lm) * ROW2B;
#pragma unroll 1
      for (int kt = 0; kt < KT2; ++kt) {
        FragAB a;
        a.u[0] = *(const v4u*)(arow + kt * 64 + half * 32);
        a.u[1] = *(const v4u*)(arow + kt * 64 + half * 32 + 16);
#pragma unroll
        for (int g = 0; g < 4; ++g) {
          FragAB b;
          const char* bp = smem + (((g * KT2 + kt) << 10) + lane * 32);
          b.u[0] = *(const v4u*)bp;
          b.u[1] = *(const v4u*)(bp + 16);
          acc[g] = __builtin_amdgcn_wmma_f32_16x16x32_bf16(false, a.bf, false, b.bf,
                                                           (short)0, acc[g], false, false);
        }
      }
      // LSTM2 pointwise (branch-free); h2(t) -> X2nxt[:,1024:2048]
#pragma unroll
      for (int r = 0; r < 8; ++r) {
        float i_ = sigm(acc[0][r]), f_ = sigm(acc[1][r]);
        float g_ = fast_tanh(acc[2][r]), o_ = sigm(acc[3][r]);
        float cn = f_ * creg[r] + i_ * g_;
        creg[r] = cn;
        __bf16 hb = (__bf16)(o_ * fast_tanh(cn));
        int b = wave * 16 + r + half * 8;
        X2nxt[(size_t)b * ROW2 + 1024 + hcol] = hb;
      }
    }
    grid_barrier(bar);
  }

  // final frame: FC(T-1) reads h2(T-1) in X2[(T)&1] == X2[0]
  if (!isG1) {
    const int w2 = w - NG1;
    if (w2 < 40 && wave == 0)
      fc_tile(w2, TT - 1, X2 /* buffer 0 */, FCsw, fc_b, out, lane);
  }
}

// ---------------- host launcher ----------------
extern "C" void kernel_launch(void* const* d_in, const int* in_sizes, int n_in,
                              void* d_out, int out_size, void* d_ws, size_t ws_size,
                              hipStream_t stream) {
  (void)in_sizes; (void)n_in; (void)out_size; (void)ws_size;
  const float* enc_out = (const float*)d_in[0];
  const float* style   = (const float*)d_in[1];
  const float* mel     = (const float*)d_in[2];
  const float* W_ih1   = (const float*)d_in[3];
  const float* W_hh1   = (const float*)d_in[4];
  const float* b_ih1   = (const float*)d_in[5];
  const float* b_hh1   = (const float*)d_in[6];
  const float* W_ih2   = (const float*)d_in[7];
  const float* W_hh2   = (const float*)d_in[8];
  const float* b_ih2   = (const float*)d_in[9];
  const float* b_hh2   = (const float*)d_in[10];
  const float* fc_w    = (const float*)d_in[11];
  const float* fc_b    = (const float*)d_in[12];

  char* ws = (char*)d_ws;
  size_t o = 0;
  unsigned int* W1sw = (unsigned int*)(ws + o); o += (size_t)W1DW * 4;
  unsigned int* W2sw = (unsigned int*)(ws + o); o += (size_t)W2DW * 4;
  unsigned int* FCsw = (unsigned int*)(ws + o); o += (size_t)FCDW * 4;
  __bf16* X1 = (__bf16*)(ws + o); o += (size_t)2 * X1BUF * 2;
  __bf16* X2 = (__bf16*)(ws + o); o += (size_t)2 * X2BUF * 2;
  int* bar = (int*)(ws + o);   // total ws use ~39.5 MB

  hipMemsetAsync(bar, 0, 256, stream);
  prep_w1<<<W1DW / 256, 256, 0, stream>>>(W_ih1, W_hh1, W1sw);
  prep_w2<<<W2DW / 256, 256, 0, stream>>>(W_ih2, W_hh2, W2sw);
  prep_fc<<<(FCDW + 255) / 256, 256, 0, stream>>>(fc_w, FCsw);

  hipFuncSetAttribute(reinterpret_cast<const void*>(decoder_persistent),
                      hipFuncAttributeMaxDynamicSharedMemorySize, LDS_BYTES);
  decoder_persistent<<<NWG, 256, LDS_BYTES, stream>>>(
      enc_out, style, mel, b_ih1, b_hh1, b_ih2, b_hh2, fc_b,
      W1sw, W2sw, FCsw, X1, X2, bar, (float*)d_out);
}
